// Decoder_19988777796283
// MI455X (gfx1250) — matrix operand order repaired and checked by reading, not compiled, not run on hardware
//
#include <hip/hip_runtime.h>
#include <hip/hip_bf16.h>

// ---------------------------------------------------------------------------
// CDNA5 (gfx1250) fused GRU decoder.
// Fusion:  gx = msg @ (W_ih@W_emb)^T + (b_ih + W_ih@b_emb)   (t>=1, K=21->32)
//          t=0 step is a per-column constant h1[j] (init_emb const, h0=0).
// All GEMMs: v_wmma_f32_16x16x32_f16, M = 16 batch rows per wave32.
// Round 3: batched B-fragment staging (pipelined s_wait_dscnt instead of
// load->wait(0)->wmma serialization); bias table in LDS (no cndmask selects).
// ---------------------------------------------------------------------------

typedef __attribute__((ext_vector_type(8)))  float     v8f;
typedef __attribute__((ext_vector_type(16))) _Float16  v16h;

union FragU { uint4 q[2]; v16h h; };

// ---- workspace layout (dword units) ---------------------------------------
// frag f occupies 256 dwords at f*256; packed per-lane: dword = lane*8 + v
//   f 0..11   : W_comb  B-frags (nt = f, kbase = 0, K padded 21->32)
//   f 12..35  : W_hh    B-frags (f = 12 + nt*2 + kf)
//   f 36..67  : W1      B-frags (f = 36 + nt*2 + kf)
//   f 68..459 : W2      B-frags (f = 68 + nt*8 + kf)
// bias block (320 f32, also copied to LDS):
//   +0   : bias_rz r (64) | z (64)  = b_ih + W_ih@b_emb (+ b_hh)
//   +128 : bias_xn (64)             = b_ih + W_ih@b_emb
//   +192 : bias_hn (64)             = b_hh (n part)
//   +256 : h1 (64)                  = hidden after step 0 (batch-const)
#define OFF_BIAS 117760

__device__ inline float fast_rcp(float x) { return __builtin_amdgcn_rcpf(x); }

__device__ inline float sigmoid_fast(float x) {
  return fast_rcp(1.0f + __expf(-x));           // v_exp + v_rcp
}
__device__ inline float tanh_fast(float x) {
  float t = __expf(-2.0f * fabsf(x));           // in (0,1]
  float r = (1.0f - t) * fast_rcp(1.0f + t);    // denom in [1,2]
  return copysignf(r, x);
}

__device__ inline v8f wmma16(v16h a, v16h b, v8f c) {
  return __builtin_amdgcn_wmma_f32_16x16x32_f16(false, a, false, b, (short)0, c,
                                                false, false);
}

// A-fragment (16x32 f16) from a row-major f16 LDS tile.
// ISA A layout: lane<16: M=lane, K 0..7 in V0-3, 16..23 in V4-7;
//               lane>=16: M=lane-16, K 8..15 / 24..31.  Each chunk = 16B.
__device__ inline v16h load_afrag(const _Float16* tile, int lane, int rowlen,
                                  int kbase) {
  int m  = lane & 15;
  int k0 = kbase + ((lane & 16) ? 8 : 0);
  FragU u;
  u.q[0] = *(const uint4*)(tile + m * rowlen + k0);
  u.q[1] = *(const uint4*)(tile + m * rowlen + k0 + 16);
  return u.h;
}

// B-fragment: per-lane contiguous 32B chunk (pre-packed by prep_frags).
__device__ inline v16h load_bfrag(const uint4* base, int f, int lane) {
  FragU u;
  const uint4* p = base + f * 64 + lane * 2;
  u.q[0] = p[0];
  u.q[1] = p[1];
  return u.h;
}

// ---------------------------------------------------------------------------
// prep kernels: pack weights into WMMA B-fragment layout, fold biases.
// ---------------------------------------------------------------------------
__device__ inline float wcomb_elem(const float* W_ih, const float* W_emb,
                                   int g, int k) {
  if (k >= 21) return 0.0f;
  float s = 0.0f;
  for (int e = 0; e < 32; ++e) s += W_ih[g * 32 + e] * W_emb[e * 21 + k];
  return s;
}

__global__ void prep_frags(const float* __restrict__ W_emb,
                           const float* __restrict__ W_ih,
                           const float* __restrict__ W_hh,
                           const float* __restrict__ W1,
                           const float* __restrict__ W2,
                           unsigned* __restrict__ wsf) {
  for (int idx = blockIdx.x * blockDim.x + threadIdx.x; idx < 117760;
       idx += gridDim.x * blockDim.x) {
    int f = idx >> 8;
    int r = idx & 255;
    int lane = r >> 3, v = r & 7;
    int ln = lane & 15;
    int k  = ((lane < 16) ? 0 : 8) + 2 * v + ((v >= 4) ? 8 : 0);   // B layout K index
    float lo, hi;
    if (f < 12) {                       // W_comb (K = vocab, padded)
      int n = f * 16 + ln;
      lo = wcomb_elem(W_ih, W_emb, n, k);
      hi = wcomb_elem(W_ih, W_emb, n, k + 1);
    } else if (f < 36) {                // W_hh^T
      int fi = f - 12, nt = fi >> 1, kf = fi & 1;
      int n = nt * 16 + ln, kk = kf * 32 + k;
      lo = W_hh[n * 64 + kk]; hi = W_hh[n * 64 + kk + 1];
    } else if (f < 68) {                // W1^T
      int fi = f - 36, nt = fi >> 1, kf = fi & 1;
      int n = nt * 16 + ln, kk = kf * 32 + k;
      lo = W1[n * 64 + kk]; hi = W1[n * 64 + kk + 1];
    } else {                            // W2^T
      int fi = f - 68, nt = fi >> 3, kf = fi & 7;
      int n = nt * 16 + ln, kk = kf * 32 + k;
      lo = W2[n * 256 + kk]; hi = W2[n * 256 + kk + 1];
    }
    union { _Float16 h[2]; unsigned u; } p;
    p.h[0] = (_Float16)lo; p.h[1] = (_Float16)hi;
    wsf[idx] = p.u;
  }
}

__global__ void prep_vectors(const float* __restrict__ init_emb,
                             const float* __restrict__ W_emb,
                             const float* __restrict__ b_emb,
                             const float* __restrict__ W_ih,
                             const float* __restrict__ b_ih,
                             const float* __restrict__ b_hh,
                             float* __restrict__ ws) {
  int g = threadIdx.x;
  if (g < 192) {
    float xb = b_ih[g];
    for (int e = 0; e < 32; ++e) xb += W_ih[g * 32 + e] * b_emb[e];
    if (g < 128) ws[OFF_BIAS + g] = xb + b_hh[g];          // rz
    else         ws[OFF_BIAS + g] = xb;                    // xn at +128
  }
  if (g < 64) {
    ws[OFF_BIAS + 192 + g] = b_hh[128 + g];                // hn
    float g0r = b_ih[g], g0z = b_ih[64 + g], g0n = b_ih[128 + g];
    for (int e = 0; e < 32; ++e) {
      float ie = init_emb[e];
      g0r += W_ih[g * 32 + e] * ie;
      g0z += W_ih[(64 + g) * 32 + e] * ie;
      g0n += W_ih[(128 + g) * 32 + e] * ie;
    }
    float rr = sigmoid_fast(g0r + b_hh[g]);
    float zz = sigmoid_fast(g0z + b_hh[64 + g]);
    float nn = tanh_fast(g0n + rr * b_hh[128 + g]);
    ws[OFF_BIAS + 256 + g] = (1.0f - zz) * nn;             // h1 (h0 = 0)
  }
}

// ---------------------------------------------------------------------------
// Main fused kernel: 8 waves/block, each wave owns 16 batch rows.
// LDS (f16 units): [0,18432)        36 weight frags (W_comb + W_hh)
//                  [18432,26624)    per-wave h buffer 16x64
//                  [26624,30720)    per-wave msg/x transpose tile 16x32
//                  [30720,31360)    bias table (320 f32)
// Total 62720 bytes dynamic LDS.
// ---------------------------------------------------------------------------
#define LB 15360   // bias table base, dword units within LDS

__global__ __launch_bounds__(256) void decoder_main(
    const float* __restrict__ msg, const int* __restrict__ lengths,
    const float* __restrict__ ws, const float* __restrict__ b1,
    const float* __restrict__ b2, float* __restrict__ out) {
  extern __shared__ _Float16 lds[];
  const int tid  = threadIdx.x;
  const int lane = tid & 31;
  const int wv   = tid >> 5;
  const int rowbase = (blockIdx.x * 8 + wv) * 16;

  {  // stage W_comb + W_hh fragments + bias table into LDS
    const unsigned* src = (const unsigned*)ws;
    unsigned* dst = (unsigned*)lds;
    for (int i = tid; i < 9216; i += 256) dst[i] = src[i];
    for (int i = tid; i < 320; i += 256) dst[LB + i] = src[OFF_BIAS + i];
  }
  __syncthreads();

  _Float16* hbuf = lds + 18432 + wv * 1024;   // 16 x 64
  _Float16* mbuf = lds + 26624 + wv * 512;    // 16 x 32
  const uint4* ldsfr = (const uint4*)lds;
  const uint4* wsfr  = (const uint4*)ws;
  const float* blds  = (const float*)lds;     // dword-indexed LDS view

  for (int i = lane; i < 512; i += 32) mbuf[i] = (_Float16)0.0f;  // K pad

  const int ln   = lane & 15;
  const int moff = (lane & 16) ? 8 : 0;

  v8f h[4], last[4];
#pragma unroll
  for (int jt = 0; jt < 4; ++jt) {
    float hv = blds[LB + 256 + jt * 16 + ln];   // step-0 hidden (batch-const)
#pragma unroll
    for (int i = 0; i < 8; ++i) { h[jt][i] = hv; last[jt][i] = hv; }
#pragma unroll
    for (int r = 0; r < 8; ++r)
      hbuf[(moff + r) * 64 + jt * 16 + ln] = (_Float16)hv;
  }
  int lenf[8];
#pragma unroll
  for (int r = 0; r < 8; ++r) lenf[r] = lengths[rowbase + moff + r];

  // ------------------------- recurrent loop: t = 1..29 --------------------
#pragma unroll 1
  for (int t = 1; t < 30; ++t) {
    const float* mrow = msg + ((long)rowbase * 30 + (t - 1)) * 21;
    for (int i = lane; i < 336; i += 32) {      // 16 rows x 21 vocab
      int m = i / 21, v = i - m * 21;
      mbuf[m * 32 + v] = (_Float16)mrow[(long)m * 630 + v];
    }
    if (t < 29) __builtin_prefetch(mrow + 21, 0, 1);  // next step's row
    asm volatile("s_wait_dscnt 0" ::: "memory");      // intra-wave LDS RAW

    v16h amsg = load_afrag(mbuf, lane, 32, 0);
    v16h ah0  = load_afrag(hbuf, lane, 64, 0);
    v16h ah1  = load_afrag(hbuf, lane, 64, 32);

    // Rolled: one jt's state live (~225 VGPRs incl. staging) => >=2 waves/SIMD.
#pragma unroll 1
    for (int jt = 0; jt < 4; ++jt) {
      // bias scalars from LDS table (no register-array cndmask selects)
      float brz_r = blds[LB +       jt * 16 + ln];
      float brz_z = blds[LB +  64 + jt * 16 + ln];
      float bxn   = blds[LB + 128 + jt * 16 + ln];
      float bhn   = blds[LB + 192 + jt * 16 + ln];
      // batch all 9 B-fragment loads -> staggered dscnt waits, not wait(0)
      v16h bcr = load_bfrag(ldsfr, jt,     lane);
      v16h bcz = load_bfrag(ldsfr, jt + 4, lane);
      v16h bcn = load_bfrag(ldsfr, jt + 8, lane);
      v16h bhr0 = load_bfrag(ldsfr, 12 + jt * 2,           lane);
      v16h bhr1 = load_bfrag(ldsfr, 12 + jt * 2 + 1,       lane);
      v16h bhz0 = load_bfrag(ldsfr, 12 + (jt + 4) * 2,     lane);
      v16h bhz1 = load_bfrag(ldsfr, 12 + (jt + 4) * 2 + 1, lane);
      v16h bhn0 = load_bfrag(ldsfr, 12 + (jt + 8) * 2,     lane);
      v16h bhn1 = load_bfrag(ldsfr, 12 + (jt + 8) * 2 + 1, lane);

      v8f cr, cz, cxn, chn;
#pragma unroll
      for (int i = 0; i < 8; ++i) {
        cr[i] = brz_r; cz[i] = brz_z; cxn[i] = bxn; chn[i] = bhn;
      }
      // interleaved accumulators: consecutive WMMAs target different D regs
      cr  = wmma16(amsg, bcr,  cr);
      cz  = wmma16(amsg, bcz,  cz);
      cxn = wmma16(amsg, bcn,  cxn);
      cr  = wmma16(ah0,  bhr0, cr);
      cz  = wmma16(ah0,  bhz0, cz);
      chn = wmma16(ah0,  bhn0, chn);
      cr  = wmma16(ah1,  bhr1, cr);
      cz  = wmma16(ah1,  bhz1, cz);
      chn = wmma16(ah1,  bhn1, chn);

      // gates, state update, length-gated capture — all in C-fragment regs
#pragma unroll
      for (int i = 0; i < 8; ++i) {
        float rg = sigmoid_fast(cr[i]);
        float zg = sigmoid_fast(cz[i]);
        float ng = tanh_fast(cxn[i] + rg * chn[i]);
        float hn2 = (1.0f - zg) * ng + zg * h[jt][i];
        h[jt][i] = hn2;
        last[jt][i] = (lenf[i] == t + 1) ? hn2 : last[jt][i];
      }
#pragma unroll
      for (int r = 0; r < 8; ++r)   // f16 copy for next step's A operand
        hbuf[(moff + r) * 64 + jt * 16 + ln] = (_Float16)h[jt][r];
    }
  }

  // ------------------------- FC1: elu(last @ W1^T + b1) -------------------
#pragma unroll
  for (int jt = 0; jt < 4; ++jt)
#pragma unroll
    for (int r = 0; r < 8; ++r)
      hbuf[(moff + r) * 64 + jt * 16 + ln] = (_Float16)last[jt][r];
  asm volatile("s_wait_dscnt 0" ::: "memory");
  v16h al0 = load_afrag(hbuf, lane, 64, 0);
  v16h al1 = load_afrag(hbuf, lane, 64, 32);

  v16h ax[8];  // x (16x256) kept resident as 8 A-fragments
#pragma unroll 1
  for (int kf = 0; kf < 8; ++kf) {
#pragma unroll
    for (int half = 0; half < 2; ++half) {
      int nt = kf * 2 + half;
      v16h bw0 = load_bfrag(wsfr, 36 + nt * 2,     lane);
      v16h bw1 = load_bfrag(wsfr, 36 + nt * 2 + 1, lane);
      float bb = b1[nt * 16 + ln];
      v8f c;
#pragma unroll
      for (int i = 0; i < 8; ++i) c[i] = bb;
      c = wmma16(al0, bw0, c);
      c = wmma16(al1, bw1, c);
#pragma unroll
      for (int r = 0; r < 8; ++r) {
        float x = c[r];
        x = x > 0.0f ? x : (__expf(x) - 1.0f);   // elu
        mbuf[(moff + r) * 32 + half * 16 + ln] = (_Float16)x;
      }
    }
    asm volatile("s_wait_dscnt 0" ::: "memory"); // transpose via 1KB tile
    ax[kf] = load_afrag(mbuf, lane, 32, 0);
  }

  // ------------------------- FC2: sigmoid(x @ W2^T + b2) ------------------
  float* orow = out + (long)rowbase * 784;
#pragma unroll 1
  for (int nt = 0; nt < 49; ++nt) {
    v16h bw[8];                                  // batch global frag loads
#pragma unroll
    for (int kf = 0; kf < 8; ++kf)
      bw[kf] = load_bfrag(wsfr, 68 + nt * 8 + kf, lane);
    float bb = b2[nt * 16 + ln];
    v8f c;
#pragma unroll
    for (int i = 0; i < 8; ++i) c[i] = bb;
#pragma unroll
    for (int kf = 0; kf < 8; ++kf)
      c = wmma16(ax[kf], bw[kf], c);
#pragma unroll
    for (int r = 0; r < 8; ++r)
      orow[(long)(moff + r) * 784 + nt * 16 + ln] = sigmoid_fast(c[r]);
  }
}

// ---------------------------------------------------------------------------
extern "C" void kernel_launch(void* const* d_in, const int* in_sizes, int n_in,
                              void* d_out, int out_size, void* d_ws,
                              size_t ws_size, hipStream_t stream) {
  const float* message  = (const float*)d_in[0];
  const int*   lengths  = (const int*)d_in[1];
  const float* init_emb = (const float*)d_in[2];
  const float* W_emb    = (const float*)d_in[3];
  const float* b_emb    = (const float*)d_in[4];
  const float* W_ih     = (const float*)d_in[5];
  const float* W_hh     = (const float*)d_in[6];
  const float* b_ih     = (const float*)d_in[7];
  const float* b_hh     = (const float*)d_in[8];
  const float* W1       = (const float*)d_in[9];
  const float* b1       = (const float*)d_in[10];
  const float* W2       = (const float*)d_in[11];
  const float* b2       = (const float*)d_in[12];
  float* out = (float*)d_out;
  float* ws  = (float*)d_ws;

  prep_frags<<<232, 256, 0, stream>>>(W_emb, W_ih, W_hh, W1, W2, (unsigned*)ws);
  prep_vectors<<<1, 256, 0, stream>>>(init_emb, W_emb, b_emb, W_ih, b_ih, b_hh, ws);
  // 65536 rows / (8 waves * 16 rows) = 512 blocks; 61.25KB dynamic LDS
  decoder_main<<<512, 256, 62720, stream>>>(message, lengths, ws, b1, b2, out);
}
